// GATSpatial_36112085025002
// MI455X (gfx1250) — matrix operand order repaired and checked by reading, not compiled, not run on hardware
//
#include <hip/hip_runtime.h>

// ---------------------------------------------------------------------------
// GAT-style double attention for MI455X (gfx1250, wave32).
// WMMA bf16->f32 flash attention, computed transposed (S^T = K Q^T) so the
// post-softmax P tile lands directly in WMMA A-operand layout (no LDS bounce).
// TDM (tensor_load_to_lds) double-buffered K tile staging via TENSORcnt.
// ---------------------------------------------------------------------------

typedef __attribute__((ext_vector_type(16))) __bf16         v16bf;
typedef __attribute__((ext_vector_type(16))) unsigned short v16us;
typedef __attribute__((ext_vector_type(8)))  float          v8f;
typedef __attribute__((ext_vector_type(4)))  unsigned int   v4u;
typedef __attribute__((ext_vector_type(8)))  int            v8i;
typedef __attribute__((ext_vector_type(4)))  int            v4i;

#define NEGV (-1.0e16f)

static __device__ __forceinline__ unsigned short f2bf(float f) {
    unsigned u = __builtin_bit_cast(unsigned, f);
    unsigned r = ((u >> 16) & 1u) + 0x7FFFu;   // round-to-nearest-even
    return (unsigned short)((u + r) >> 16);
}
static __device__ __forceinline__ float bf2f(unsigned short s) {
    return __builtin_bit_cast(float, ((unsigned)s) << 16);
}

static __device__ __forceinline__ v8f wmma_bf16(v16bf a, v16bf b, v8f c) {
    return __builtin_amdgcn_wmma_f32_16x16x32_bf16(
        /*neg_a=*/false, a, /*neg_b=*/false, b,
        /*c_mod=*/(short)0, c, /*reuse_a=*/false, /*reuse_b=*/false);
}

// A-operand (16x32 bf16) gather per ISA layout (two 16B contiguous runs):
// lanes 0-15: row M=lane, elems cover K {kb+0..7, kb+16..23}
// lanes16-31: row M=lane-16, elems cover K {kb+8..15, kb+24..31}
static __device__ __forceinline__ v16bf loadA(const unsigned short* row, int kb, int hi) {
    v16us u;
#pragma unroll
    for (int e = 0; e < 16; ++e) {
        int k = kb + ((e < 8) ? e : (e + 8)) + 8 * hi;
        u[e] = row[k];
    }
    return __builtin_bit_cast(v16bf, u);
}

// B-operand (32x16 bf16): lane's k-elements {p+0..7, p+16..23} for its column.
static __device__ __forceinline__ v16bf loadB(const unsigned short* p) {
    v16us u;
#pragma unroll
    for (int e = 0; e < 16; ++e) u[e] = p[(e < 8) ? e : (e + 8)];
    return __builtin_bit_cast(v16bf, u);
}

// Pack two 8-float register strips into one bf16 A-operand (pure per-lane).
static __device__ __forceinline__ v16bf packA(const float* p0, const float* p1) {
    v16us u;
#pragma unroll
    for (int e = 0; e < 8; ++e) { u[e] = f2bf(p0[e]); u[e + 8] = f2bf(p1[e]); }
    return __builtin_bit_cast(v16bf, u);
}

static __device__ __forceinline__ float redsum16(float v) {
#pragma unroll
    for (int m = 1; m < 16; m <<= 1) v += __shfl_xor(v, m, 32);
    return v;
}

// ---------------------------------------------------------------------------
// TDM: DMA one 64x64 bf16 tile (row-major) global -> LDS at byte offset
// lds_off, padding 4 DWORDs after each 32-DWORD row => row stride 72 bf16.
// amdgpu-toolchain (clang-23) builtin: 6 args (extra int32x8 group zeroed).
// ---------------------------------------------------------------------------
static __device__ __forceinline__ void tdm_load_tile(const unsigned short* gsrc,
                                                     unsigned lds_off) {
    unsigned long long ga = (unsigned long long)(size_t)gsrc;
    v4u g0;
    g0[0] = 1u;                                   // count=1 (valid user D#)
    g0[1] = lds_off;                              // lds_addr (bytes)
    g0[2] = (unsigned)(ga & 0xFFFFFFFFu);         // global_addr[31:0]
    g0[3] = (unsigned)((ga >> 32) & 0x01FFFFFFu)  // global_addr[56:32]
          | (2u << 30);                           // type = 2 ("image")
    v8i g1;
    g1[0] = (1 << 16)                             // data_size: 2 bytes
          | (1 << 20)                             // pad_enable
          | (4 << 22)                             // pad_interval: 32 DWORDs
          | (3 << 25);                            // pad_amount: 4 DWORDs
    g1[1] = (64 << 16);                           // tensor_dim0 = 64 (lo16)
    g1[2] = (64 << 16);                           // dim0 hi=0 | tensor_dim1 = 64
    g1[3] = (64 << 16);                           // dim1 hi=0 | tile_dim0 = 64
    g1[4] = 64;                                   // tile_dim1 = 64, tile_dim2 = 0
    g1[5] = 64;                                   // tensor_dim0_stride = 64
    g1[6] = 0;
    g1[7] = 0;
    v4i z4 = {0, 0, 0, 0};
    v8i z8 = {0, 0, 0, 0, 0, 0, 0, 0};
    __builtin_amdgcn_tensor_load_to_lds(g0, g1, z4, z4, z8, 0);
}

// ---------------------------------------------------------------------------
// Kernel 0: pack (graph + I) > 0 into row-major bitmask.
// maskR[i*(N/32) + j/32] bit (j%32) == mask[row=i][col=j]; contiguous reads.
// ---------------------------------------------------------------------------
__global__ __launch_bounds__(256) void pack_mask_k(const int* __restrict__ graph,
                                                   unsigned* __restrict__ maskR) {
    const int N = 2048;
    int t = blockIdx.x * 256 + threadIdx.x;
    int i = t >> 6;
    int j32 = t & 63;
    size_t base = (size_t)i * N + j32 * 32;
    unsigned w = 0;
#pragma unroll 4
    for (int b = 0; b < 32; ++b) {
        bool m = (graph[base + b] != 0) || (i == j32 * 32 + b);
        w |= (m ? 1u : 0u) << b;
    }
    maskR[(size_t)i * 64 + j32] = w;
}

// ---------------------------------------------------------------------------
// Kernel 1: per-head projection  h[b,h,n,d] = sum_c x[b,n,c] * Wh[h,d,c]
// ---------------------------------------------------------------------------
__global__ __launch_bounds__(256) void proj1_k(const float* __restrict__ x,
                                               const float* __restrict__ Wh,
                                               unsigned short* __restrict__ hbf) {
    const int N = 2048;
    int bn = blockIdx.x;
    __shared__ float xr[64];
    int tid = threadIdx.x;
    if (tid < 64) xr[tid] = x[(size_t)bn * 64 + tid];
    __syncthreads();
    int h = tid >> 6, d = tid & 63;
    const float* w = Wh + ((size_t)(h * 64 + d)) * 64;
    float acc = 0.f;
#pragma unroll
    for (int c = 0; c < 64; ++c) acc = fmaf(xr[c], w[c], acc);
    int b = bn >> 11, n = bn & (N - 1);
    hbf[(((size_t)(b * 4 + h) * N) + n) * 64 + d] = f2bf(acc);
}

// ---------------------------------------------------------------------------
// Kernel 3: h2[b,n,d] = sum_k z[b,n,k] * Wo[d,k]   (z already leaky-relu'd)
// ---------------------------------------------------------------------------
__global__ __launch_bounds__(64) void proj2_k(const unsigned short* __restrict__ z,
                                              const float* __restrict__ Wo,
                                              unsigned short* __restrict__ h2) {
    int bn = blockIdx.x;
    __shared__ float zr[256];
    int tid = threadIdx.x;
#pragma unroll
    for (int i = tid; i < 256; i += 64) zr[i] = bf2f(z[(size_t)bn * 256 + i]);
    __syncthreads();
    const float* w = Wo + (size_t)tid * 256;
    float acc = 0.f;
#pragma unroll 8
    for (int k = 0; k < 256; ++k) acc = fmaf(zr[k], w[k], acc);
    h2[(size_t)bn * 64 + tid] = f2bf(acc);
}

// ---------------------------------------------------------------------------
// Fused masked flash-attention:  O = softmax(Q Q^T + mask) Q  (+bias, act).
// Computed transposed: S^T = K Q^T, so lane = q-row m (softmax stats are
// per-lane scalars) and post-softmax P registers ARE the A-operand for P*V.
// MODE 0: layer-1 -> leaky-relu(O + bh) bf16 into cat buffer [B,N,256].
// MODE 1: layer-2 -> leaky-relu + LayerNorm fused, fp32 to d_out [B,N,64].
// Block = 128 threads (4 waves); wave owns 16 q rows; grid = (N/64, G).
// ---------------------------------------------------------------------------
template <int MODE>
__global__ __launch_bounds__(128) void gat_attn_k(const unsigned short* __restrict__ qk,
                                                  const unsigned* __restrict__ maskR,
                                                  const float* __restrict__ bias,
                                                  const float* __restrict__ gamma,
                                                  const float* __restrict__ beta,
                                                  unsigned short* __restrict__ zout,
                                                  float* __restrict__ fout,
                                                  int H) {
    const int N  = 2048;
    const int NT = N / 64;
    const int KS = 72;  // padded LDS row stride (bf16): 144 B, 16B aligned
    __shared__ __align__(16) unsigned short Kn[2][64 * KS];  // K tiles [n][d], dbl-buf
    __shared__ __align__(16) unsigned short Vt[64 * KS];     // V tile transposed [d][n]

    const int tid  = threadIdx.x;
    const int wave = tid >> 5;
    const int lane = tid & 31;
    const int l15  = lane & 15;
    const int hi   = lane >> 4;

    const int g     = blockIdx.y;           // b*H + h (MODE 0) or b (MODE 1)
    const int head  = g % H;
    const int qbase = blockIdx.x * 64 + wave * 16;
    const int qm    = qbase + l15;          // this lane's q row (softmax row)

    // Q^T as two persistent B-operands (k = d 0..31 / 32..63).
    const unsigned short* Qrow = qk + ((size_t)g * N + qm) * 64;
    v16bf bq0 = loadB(Qrow + 8 * hi);
    v16bf bq1 = loadB(Qrow + 32 + 8 * hi);

    v8f o[4];
#pragma unroll
    for (int ct = 0; ct < 4; ++ct) o[ct] = {};
    float m_run = NEGV, l_run = 0.f;        // per-lane scalars (row = qm)

    const unsigned short* src_base = qk + (size_t)g * N * 64;
    const unsigned* mrow = maskR + (size_t)qm * 64;

    if (wave == 0) {                        // prologue DMA: tile 0 -> buffer 0
        tdm_load_tile(src_base, (unsigned)(size_t)(const void*)&Kn[0][0]);
        __builtin_amdgcn_s_wait_tensorcnt(0);
    }
    __syncthreads();

    for (int jt = 0; jt < NT; ++jt) {
        const int cur = jt & 1;
        const unsigned short* Kc = &Kn[cur][0];

        if (wave == 0 && jt + 1 < NT)       // DMA next tile (overlaps compute)
            tdm_load_tile(src_base + (jt + 1) * 64 * 64,
                          (unsigned)(size_t)(const void*)&Kn[cur ^ 1][0]);
        if (jt + 2 < NT)                    // warm GL2 for the DMA after that
            __builtin_prefetch(src_base + (jt + 2) * 64 * 64 + tid * 32, 0, 0);

        // ---- build transposed V tile from LDS K tile ----
#pragma unroll
        for (int c = 0; c < 4; ++c) {
            int chunk = tid + c * 128;      // 0..511
            int n = chunk >> 3, d8 = (chunk & 7) * 8;
            uint4 v = *(const uint4*)(Kc + n * KS + d8);
            unsigned short e[8];
            __builtin_memcpy(e, &v, 16);
#pragma unroll
            for (int i = 0; i < 8; ++i) Vt[(d8 + i) * KS + n] = e[i];
        }

        // ---- S^T = K Q^T : 4 n-tiles x 2 WMMA; lane = q-row, VGPR = n ----
        uint2 mw = *(const uint2*)(mrow + jt * 2);  // 64 mask bits of this q-row
        float s[4][8];
#pragma unroll
        for (int ct = 0; ct < 4; ++ct) {
            const unsigned short* krow = Kc + (ct * 16 + l15) * KS;
            v8f acc = {};
            acc = wmma_bf16(loadA(krow, 0, hi), bq0, acc);
            acc = wmma_bf16(loadA(krow, 32, hi), bq1, acc);
            unsigned w = (ct < 2) ? mw.x : mw.y;
            int bitbase = 16 * (ct & 1) + 8 * hi;   // n-in-tile = 16ct + r + 8hi
#pragma unroll
            for (int r = 0; r < 8; ++r)
                s[ct][r] = ((w >> (bitbase + r)) & 1u) ? acc[r] : NEGV;
        }

        __syncthreads();  // Vt fully built before any wave's PV consumes it

        // ---- flash softmax (per-lane row stats; one cross-half shfl) ----
        float mt = s[0][0];
#pragma unroll
        for (int ct = 0; ct < 4; ++ct)
#pragma unroll
            for (int r = 0; r < 8; ++r) mt = fmaxf(mt, s[ct][r]);
        mt = fmaxf(mt, __shfl_xor(mt, 16, 32));
        float mnew  = fmaxf(m_run, mt);
        float scl   = __expf(m_run - mnew);
        m_run = mnew;

        float p[4][8];
        float psum = 0.f;
#pragma unroll
        for (int ct = 0; ct < 4; ++ct)
#pragma unroll
            for (int r = 0; r < 8; ++r) {
                float sv = s[ct][r];
                float pv = (sv <= 0.5f * NEGV) ? 0.f : __expf(sv - mnew);
                p[ct][r] = pv;
                psum += pv;
            }
        psum += __shfl_xor(psum, 16, 32);
        l_run = l_run * scl + psum;

        // rescale O: row of o[ct][r] is m = r + 8hi -> fetch that row's scale
        float sc[8];
#pragma unroll
        for (int r = 0; r < 8; ++r) sc[r] = __shfl(scl, r + 8 * hi, 32);
#pragma unroll
        for (int ct = 0; ct < 4; ++ct)
#pragma unroll
            for (int r = 0; r < 8; ++r) o[ct][r] *= sc[r];

        // ---- O += P V : P registers already in A-operand layout ----
        v16bf ap0 = packA(p[0], p[1]);   // k = n 0..31
        v16bf ap1 = packA(p[2], p[3]);   // k = n 32..63
#pragma unroll
        for (int ct = 0; ct < 4; ++ct) {
            const unsigned short* vb = &Vt[(ct * 16 + l15) * KS + 8 * hi];
            o[ct] = wmma_bf16(ap0, loadB(vb), o[ct]);
            o[ct] = wmma_bf16(ap1, loadB(vb + 32), o[ct]);
        }

        if (wave == 0) __builtin_amdgcn_s_wait_tensorcnt(0);
        __syncthreads();  // next-tile DMA done; Vt free for rebuild
    }

    // ---- epilogue: redistribute 1/l from softmax-row lanes to O layout ----
    float inv[8];
#pragma unroll
    for (int r = 0; r < 8; ++r) inv[r] = 1.0f / __shfl(l_run, r + 8 * hi, 32);

    if constexpr (MODE == 0) {
        int b = g / H;
#pragma unroll
        for (int ct = 0; ct < 4; ++ct) {
            int d = ct * 16 + l15;
            float bv = bias[head * 64 + d];
#pragma unroll
            for (int r = 0; r < 8; ++r) {
                int row = qbase + r + 8 * hi;
                float v = o[ct][r] * inv[r] + bv;
                v = (v > 0.f) ? v : 0.2f * v;  // leaky_relu(cat)
                zout[((size_t)b * N + row) * 256 + head * 64 + d] = f2bf(v);
            }
        }
    } else {
        float vals[4][8];
#pragma unroll
        for (int ct = 0; ct < 4; ++ct) {
            float bv = bias[ct * 16 + l15];
#pragma unroll
            for (int r = 0; r < 8; ++r) {
                float v = o[ct][r] * inv[r] + bv;
                vals[ct][r] = (v > 0.f) ? v : 0.2f * v;
            }
        }
        float mu[8], rstd[8];
#pragma unroll
        for (int r = 0; r < 8; ++r) {
            float s1 = 0.f, s2 = 0.f;
#pragma unroll
            for (int ct = 0; ct < 4; ++ct) {
                float v = vals[ct][r];
                s1 += v; s2 += v * v;
            }
            s1 = redsum16(s1) * (1.0f / 64.0f);
            s2 = redsum16(s2) * (1.0f / 64.0f);
            mu[r]   = s1;
            rstd[r] = rsqrtf(s2 - s1 * s1 + 1e-5f);
        }
#pragma unroll
        for (int ct = 0; ct < 4; ++ct) {
            int d = ct * 16 + l15;
            float gm = gamma[d], bt = beta[d];
#pragma unroll
            for (int r = 0; r < 8; ++r) {
                int row = qbase + r + 8 * hi;
                fout[((size_t)g * N + row) * 64 + d] =
                    (vals[ct][r] - mu[r]) * rstd[r] * gm + bt;
            }
        }
    }
}

// ---------------------------------------------------------------------------
extern "C" void kernel_launch(void* const* d_in, const int* in_sizes, int n_in,
                              void* d_out, int out_size, void* d_ws, size_t ws_size,
                              hipStream_t stream) {
    (void)in_sizes; (void)n_in; (void)out_size; (void)ws_size;
    const int B = 8, N = 2048, H = 4;

    const float* x     = (const float*)d_in[0];
    const int*   graph = (const int*)d_in[1];
    const float* Wh    = (const float*)d_in[2];
    const float* bh    = (const float*)d_in[3];
    const float* Wo    = (const float*)d_in[4];
    const float* bo    = (const float*)d_in[5];
    const float* gamma = (const float*)d_in[6];
    const float* beta  = (const float*)d_in[7];

    char* ws = (char*)d_ws;
    size_t off = 0;
    unsigned*       maskR = (unsigned*)(ws + off);       off += (size_t)N * 64 * 4;         // 512 KB
    unsigned short* hbf   = (unsigned short*)(ws + off); off += (size_t)B * H * N * 64 * 2;  // 8 MB
    unsigned short* zcat  = (unsigned short*)(ws + off); off += (size_t)B * N * 256 * 2;     // 8 MB
    unsigned short* h2bf  = (unsigned short*)(ws + off); off += (size_t)B * N * 64 * 2;      // 2 MB

    pack_mask_k<<<(N * 64) / 256, 256, 0, stream>>>(graph, maskR);
    proj1_k<<<B * N, 256, 0, stream>>>(x, Wh, hbf);
    gat_attn_k<0><<<dim3(N / 64, B * H), 128, 0, stream>>>(
        hbf, maskR, bh, nullptr, nullptr, zcat, nullptr, H);
    proj2_k<<<B * N, 64, 0, stream>>>(zcat, Wo, h2bf);
    gat_attn_k<1><<<dim3(N / 64, B), 128, 0, stream>>>(
        h2bf, maskR, bo, gamma, beta, nullptr, (float*)d_out, 1);
}
